// PerturbationDimSelector_73315091743546
// MI455X (gfx1250) — matrix operand order repaired and checked by reading, not compiled, not run on hardware
//
#include <hip/hip_runtime.h>
#include <hip/hip_bf16.h>

// Problem constants (from reference): B=8, S=1024, H=4096, HD=32, K=512
#define BATCH 8
#define SEQ   1024
#define HDIM  4096
#define HID   32
#define KSEL  512
#define NROWS (BATCH * SEQ)          // 8192

typedef __attribute__((ext_vector_type(16))) _Float16 v16h;
typedef __attribute__((ext_vector_type(8)))  float    v8f;

// ---- CDNA5 async global->LDS helpers (ASYNCcnt-tracked DMA) -------------
// LDS byte offset = low 32 bits of the generic (flat) address of a __shared__
// object (flat LDS aperture).
__device__ __forceinline__ unsigned ldsOffset(const void* p) {
    return (unsigned)(unsigned long long)(size_t)p;
}
__device__ __forceinline__ void async_load16(unsigned ldsOff, const void* g) {
    asm volatile("global_load_async_to_lds_b128 %0, %1, off"
                 :: "v"(ldsOff), "v"((unsigned long long)(size_t)g)
                 : "memory");
}
__device__ __forceinline__ void wait_async0() {
    asm volatile("s_wait_asynccnt 0" ::: "memory");
}

// float -> order-preserving uint (ascending); top-k == largest keys
__device__ __forceinline__ unsigned orderKey(float f) {
    unsigned k = __float_as_uint(f);
    return (k & 0x80000000u) ? ~k : (k | 0x80000000u);
}

__device__ __forceinline__ unsigned makeKey(float score, float uv) {
    const float eps = 1e-6f;
    float uc = fminf(fmaxf(uv, eps), 1.0f - eps);
    float g  = -__logf(-__logf(uc));         // gumbel
    return orderKey(score + g);
}

// ---------------------------------------------------------------------------
// Kernel 1: h = relu(X @ W1 + b1)   (M=8192, K=4096, N=32) via WMMA f16->f32
// grid = 512 blocks (16 rows each), block = 64 threads (2 waves; wave w does
// N-tile w). X/W1 k-tiles are double-buffered in LDS via async DMA
// (global_load_async_to_lds_b128), overlapping with WMMA compute.
// Block 0 also zero-inits the 8 dim_log_probs accumulators.
// ---------------------------------------------------------------------------
__global__ __launch_bounds__(64)
void pds_gemm1_relu(const float* __restrict__ X, const float* __restrict__ W1,
                    const float* __restrict__ b1, float* __restrict__ h,
                    float* __restrict__ lpAcc) {
    // double-buffered staging tiles (12 KB total)
    __shared__ __align__(16) float xs[2][16 * 32];   // X tile: 16 rows x 32 k
    __shared__ __align__(16) float ws[2][32 * 32];   // W1 tile: 32 k x 32 n

    const int tid = threadIdx.x;
    if (blockIdx.x == 0 && tid < BATCH) lpAcc[tid] = 0.0f;

    const int rowBase = blockIdx.x * 16;
    const int lane    = tid & 31;
    const int wave    = tid >> 5;       // 0..1 -> N tile
    const int m       = lane & 15;      // A row within tile
    const int hi      = lane >> 4;      // lane-half selector
    const int n       = lane & 15;      // B/C column within tile
    const int colBase = wave * 16;
    const int klo     = hi * 16;        // unused; B K-offset is hi*8

    // stage one k-chunk: all 64 threads issue 16B async chunks
    auto stage = [&](int buf, int k0) {
        // X tile: 16 rows x 128B  (128 chunks of 16B)
        #pragma unroll
        for (int rep = 0; rep < 2; ++rep) {
            const int cc   = tid + rep * 64;         // 0..127
            const int row  = cc >> 3, part = cc & 7; // part = 16B chunk in row
            const float* g = X + (size_t)(rowBase + row) * HDIM + k0 + part * 4;
            async_load16(ldsOffset(&xs[buf][cc * 4]), g);
        }
        // W1 tile: 32 rows x 128B  (256 chunks of 16B)
        #pragma unroll
        for (int rep = 0; rep < 4; ++rep) {
            const int cc   = tid + rep * 64;         // 0..255
            const int krow = cc >> 3, part = cc & 7;
            const float* g = W1 + (size_t)(k0 + krow) * HID + part * 4;
            async_load16(ldsOffset(&ws[buf][cc * 4]), g);
        }
    };

    v8f c = {};
    stage(0, 0);
    int cur = 0;

    for (int k0 = 0; k0 < HDIM; k0 += 32) {
        wait_async0();          // my batch for buf[cur] has landed
        __syncthreads();        // all waves' batches landed; old consumers done
        if (k0 + 32 < HDIM) stage(cur ^ 1, k0 + 32);   // DMA next chunk now

        // A fragment: 16x32 f16 from LDS. halves 0..7 -> K=hi*8+o,
        // halves 8..15 -> K=16+hi*8+o (per ISA 16-bit A layout)
        v16h a;
        const float* xa = &xs[cur][m * 32 + hi * 8];
        #pragma unroll
        for (int o = 0; o < 8; ++o) {
            a[o]     = (_Float16)xa[o];
            a[8 + o] = (_Float16)xa[16 + o];
        }
        // B fragment: 32x16 f16 from LDS W1 tile. half d -> K = hi*8 + d + (d & 8)
        v16h b;
        const float* wc = &ws[cur][hi * 8 * 32 + colBase + n];
        #pragma unroll
        for (int d = 0; d < 16; ++d) b[d] = (_Float16)wc[(d + (d & 8)) * 32];

        c = __builtin_amdgcn_wmma_f32_16x16x32_f16(
                false, a, false, b, (short)0, c, false, false);
        cur ^= 1;
    }

    const float bias = b1[colBase + n];
    #pragma unroll
    for (int i = 0; i < 8; ++i) {
        const int mm = i + 8 * hi;      // C layout: VGPR i -> M = i + 8*(lane/16)
        float v = c[i] + bias;
        h[(size_t)(rowBase + mm) * HID + colBase + n] = v > 0.0f ? v : 0.0f;
    }
}

// ---------------------------------------------------------------------------
// Kernel 2: fused  scores = h @ W2 + b2 (WMMA, scores stay in 256KB LDS),
// row max / logsumexp, per-row 4-pass radix top-512 over score+gumbel keys,
// index emission, and dim_log_probs accumulation.
// grid = 512 blocks (16 rows each), block = 256 threads (8 waves), dynamic
// LDS = 16*4096*4 = 262144 bytes.
// ---------------------------------------------------------------------------
__global__ __launch_bounds__(256)
void pds_scores_topk(const float* __restrict__ h,  const float* __restrict__ W2,
                     const float* __restrict__ b2, const float* __restrict__ u,
                     int* __restrict__ outIdx,     float* __restrict__ lpAcc) {
    extern __shared__ float sc[];                 // [16][HDIM] score tile
    __shared__ unsigned hist[256];
    __shared__ float rowMax[16], rowLse[16];
    __shared__ float red[16][17];                 // also reused flat (272 >= 256)
    __shared__ unsigned ctl[8];                   // 0:prefix 1:kRem 2:posCnt 3:tieCnt

    const int tid     = threadIdx.x;
    const int wave    = tid >> 5;                 // 0..7
    const int lane    = tid & 31;
    const int rowBase = blockIdx.x * 16;

    // ---- GEMM2 tile: A = h[rowBase:rowBase+16, 0:32] as f16 fragment ----
    const int m  = lane & 15;
    const int hi = lane >> 4;
    const int n  = lane & 15;
    const int klo = hi * 16;
    v16h a;
    {
        const float* hr = h + (size_t)(rowBase + m) * HID + hi * 8;
        #pragma unroll
        for (int o = 0; o < 8; ++o) {
            a[o]     = (_Float16)hr[o];
            a[8 + o] = (_Float16)hr[16 + o];
        }
    }
    // each wave covers 512 columns = 32 N-tiles
    for (int t = 0; t < 32; ++t) {
        const int j0  = wave * 512 + t * 16;
        const int col = j0 + n;
        v16h b;
        const float* wc = W2 + (size_t)(hi * 8) * HDIM + col;   // W2 is HID x HDIM
        #pragma unroll
        for (int d = 0; d < 16; ++d) b[d] = (_Float16)wc[(d + (d & 8)) * HDIM];
        v8f c = {};
        c = __builtin_amdgcn_wmma_f32_16x16x32_f16(
                false, a, false, b, (short)0, c, false, false);
        const float bias = b2[col];
        #pragma unroll
        for (int i = 0; i < 8; ++i) {
            const int mm = i + 8 * hi;
            sc[mm * HDIM + col] = c[i] + bias;
        }
    }
    __syncthreads();

    // ---- row statistics: max and logsumexp (16 threads per row) ----
    {
        const int r = tid >> 4, l = tid & 15;
        float mx = -3.4e38f;
        for (int j = l; j < HDIM; j += 16) mx = fmaxf(mx, sc[r * HDIM + j]);
        red[r][l] = mx;
        __syncthreads();
        if (l == 0) {
            float rm = red[r][0];
            #pragma unroll
            for (int q = 1; q < 16; ++q) rm = fmaxf(rm, red[r][q]);
            rowMax[r] = rm;
        }
        __syncthreads();
        const float rm = rowMax[r];
        float se = 0.0f;
        for (int j = l; j < HDIM; j += 16) se += __expf(sc[r * HDIM + j] - rm);
        red[r][l] = se;
        __syncthreads();
        if (l == 0) {
            float s = 0.0f;
            #pragma unroll
            for (int q = 0; q < 16; ++q) s += red[r][q];
            rowLse[r] = __logf(s);
        }
        __syncthreads();
    }

    // ---- per-row top-KSEL via MSB-first 8-bit radix select ----
    for (int r = 0; r < 16; ++r) {
        const int grow = rowBase + r;
        const float* urow = u + (size_t)grow * HDIM;
        if (r + 1 < 16)   // hint next row's gumbel source into cache
            __builtin_prefetch(u + (size_t)(grow + 1) * HDIM + (tid << 4), 0, 3);

        if (tid == 0) { ctl[0] = 0u; ctl[1] = KSEL; }

        for (int p = 0; p < 4; ++p) {
            hist[tid] = 0u;
            __syncthreads();
            const unsigned prefix = ctl[0];
            const int shift = 24 - 8 * p;
            for (int j = tid; j < HDIM; j += 256) {
                const unsigned key = makeKey(sc[r * HDIM + j], urow[j]);
                if (p == 0 || (key >> (shift + 8)) == prefix)
                    atomicAdd(&hist[(key >> shift) & 255u], 1u);
            }
            __syncthreads();
            if (tid == 0) {
                unsigned kRem = ctl[1], cum = 0;
                int d = 255;
                for (; d >= 0; --d) {
                    const unsigned cc = hist[d];
                    if (cum + cc >= kRem) break;
                    cum += cc;
                }
                if (d < 0) d = 0;
                ctl[0] = (prefix << 8) | (unsigned)d;
                ctl[1] = kRem - cum;
            }
            __syncthreads();
        }

        const unsigned T          = ctl[0];      // exact 32-bit threshold key
        const unsigned tiesNeeded = ctl[1];      // 1..KSEL of key==T to take
        __syncthreads();
        if (tid == 0) { ctl[2] = 0u; ctl[3] = 0u; }
        __syncthreads();

        float selSum = 0.0f;
        int* rowOut = outIdx + (size_t)grow * KSEL;
        for (int j = tid; j < HDIM; j += 256) {
            const float s = sc[r * HDIM + j];
            const unsigned key = makeKey(s, urow[j]);
            if (key > T) {
                const unsigned pos = atomicAdd(&ctl[2], 1u);
                rowOut[pos] = j;
                selSum += s;
            } else if (key == T) {
                const unsigned tp = atomicAdd(&ctl[3], 1u);
                if (tp < tiesNeeded) {
                    rowOut[(KSEL - tiesNeeded) + tp] = j;
                    selSum += s;
                }
            }
        }

        // reduce selected-score sum across block
        __syncthreads();
        float* flat = (float*)red;
        flat[tid] = selSum;
        __syncthreads();
        for (int s2 = 128; s2 > 0; s2 >>= 1) {
            if (tid < s2) flat[tid] += flat[tid + s2];
            __syncthreads();
        }
        if (tid == 0) {
            const float total = flat[0];
            // sum over selected of (score - max - logZ)
            const float lp = total - (float)KSEL * (rowMax[r] + rowLse[r]);
            const int bidx = grow / SEQ;
            atomicAdd(&lpAcc[bidx], lp * (1.0f / ((float)SEQ * (float)KSEL)));
        }
        __syncthreads();
    }
}

// ---------------------------------------------------------------------------
extern "C" void kernel_launch(void* const* d_in, const int* in_sizes, int n_in,
                              void* d_out, int out_size, void* d_ws, size_t ws_size,
                              hipStream_t stream) {
    const float* X  = (const float*)d_in[0];   // (B,S,H)
    const float* W1 = (const float*)d_in[1];   // (H,HD)
    const float* b1 = (const float*)d_in[2];   // (HD,)
    const float* W2 = (const float*)d_in[3];   // (HD,H)
    const float* b2 = (const float*)d_in[4];   // (H,)
    const float* u  = (const float*)d_in[5];   // (B,S,H)
    // d_in[6] = num_perturb_dims (K=512, compile-time constant here)

    float* hbuf   = (float*)d_ws;                                   // 8192*32 f32 = 1MB
    int*   outIdx = (int*)d_out;                                    // (B,S,K) int32 bits
    float* lpAcc  = (float*)d_out + (size_t)BATCH * SEQ * KSEL;     // (B,) f32

    // 256KB dynamic LDS for the 16x4096 f32 score tile (WGP has 320KB)
    hipFuncSetAttribute(reinterpret_cast<const void*>(pds_scores_topk),
                        hipFuncAttributeMaxDynamicSharedMemorySize,
                        16 * HDIM * (int)sizeof(float));

    pds_gemm1_relu<<<NROWS / 16, 64, 0, stream>>>(X, W1, b1, hbuf, lpAcc);
    pds_scores_topk<<<NROWS / 16, 256, 16 * HDIM * sizeof(float), stream>>>(
        hbuf, W2, b2, u, outIdx, lpAcc);
}
